// Decoder_35648228557245
// MI455X (gfx1250) — matrix operand order repaired and checked by reading, not compiled, not run on hardware
//
#include <hip/hip_runtime.h>
#include <hip/hip_bf16.h>

// ---------------------------------------------------------------------------
// Shapes (fixed by the reference)
// ---------------------------------------------------------------------------
#define BB 128   // batch
#define LL 512   // seq len
#define VV 2048  // vocab
#define HH 512   // hidden
// rows of the big GEMM = L*B = 65536

typedef __attribute__((ext_vector_type(16))) _Float16 v16h;
typedef __attribute__((ext_vector_type(8)))  _Float16 v8h;
typedef __attribute__((ext_vector_type(8)))  float    v8f;

// Hardware tanh (V_TANH_F32) on gfx1250; libm fallback keeps compile safe.
__device__ __forceinline__ float fast_tanh(float x) {
#if __has_builtin(__builtin_amdgcn_tanhf)
  return __builtin_amdgcn_tanhf(x);
#else
  return tanhf(x);
#endif
}

// sigmoid(x) = 0.5 * (1 + tanh(x/2)) -> one TRANS op + FMA
__device__ __forceinline__ float fast_sigmoid(float x) {
  return fmaf(fast_tanh(x * 0.5f), 0.5f, 0.5f);
}

// Convert 8 contiguous fp32 -> f16 into fragment elements [e..e+7]
__device__ __forceinline__ void load8f(const float* __restrict__ p, v16h& f, int e) {
  float4 a = *reinterpret_cast<const float4*>(p);
  float4 b = *reinterpret_cast<const float4*>(p + 4);
  f[e+0]=(_Float16)a.x; f[e+1]=(_Float16)a.y; f[e+2]=(_Float16)a.z; f[e+3]=(_Float16)a.w;
  f[e+4]=(_Float16)b.x; f[e+5]=(_Float16)b.y; f[e+6]=(_Float16)b.z; f[e+7]=(_Float16)b.w;
}

__device__ __forceinline__ v8f wmma16(v16h a, v16h b, v8f c) {
  // D = A(16x32 f16) * B(32x16 f16) + C(16x16 f32)
  return __builtin_amdgcn_wmma_f32_16x16x32_f16(false, a, false, b, (short)0, c,
                                                false, false);
}

// ---------------------------------------------------------------------------
// Generic GEMM:  C[M,N] = act( A[M,K] @ W[N,K]^T + bias[N] )
// A row-major stride lda, W row-major stride ldw (torch Linear weight layout).
// One 16x16 output tile per wave; K multiple of 32, M,N multiples of 16.
// ---------------------------------------------------------------------------
__global__ void k_gemm(const float* __restrict__ A, int lda,
                       const float* __restrict__ W, int ldw,
                       const float* __restrict__ bias,
                       float* __restrict__ C, int ldc,
                       int M, int N, int K, int act) {
  const int ntn  = N >> 4;
  const int tile = blockIdx.x * (blockDim.x >> 5) + (threadIdx.x >> 5);
  if (tile >= (M >> 4) * ntn) return;           // wave-uniform exit
  const int m0   = (tile / ntn) << 4;
  const int n0   = (tile % ntn) << 4;
  const int lane = threadIdx.x & 31;
  const int half = lane >> 4;
  const int l16  = lane & 15;

  const float* arow = A + (size_t)(m0 + l16) * lda + (half << 3);  // +0 or +8
  const float* wrow = W + (size_t)(n0 + l16) * ldw + (half << 3);  // +0 or +8

  v8f acc = {};
  for (int kb = 0; kb < K; kb += 32) {
    v16h af, bf;
    load8f(arow + kb,      af, 0);   // K = kb+akk .. +7
    load8f(arow + kb + 16, af, 8);   // K = kb+akk+16 .. +23
    load8f(wrow + kb,      bf, 0);   // K = kb+bkk .. +7
    load8f(wrow + kb + 16, bf, 8);   // K = kb+bkk+16 .. +23
    acc = wmma16(af, bf, acc);
  }
#pragma unroll
  for (int v = 0; v < 8; ++v) {
    int m = m0 + v + (half << 3);
    int n = n0 + l16;
    float val = acc[v] + bias[n];
    if (act) val = fast_tanh(val);
    C[(size_t)m * ldc + n] = val;
  }
}

// ---------------------------------------------------------------------------
// Fused attention-score kernel.
// Rows r = l*B + b (65536 total).  Per row:
//   e[r] = sum_n score_W[n] * tanh( qp[b,n] + sum_k enc[r,k]*attn_W[n, H+k] )
// Block: 32 rows, 256 threads (8 waves).  A tile (32x512) staged in LDS as f16.
// Wave w owns n-tiles {4w..4w+3}; 2 m-tiles each -> 8 accumulators.
// ---------------------------------------------------------------------------
#define ASTRIDE 520  // 512 + 8 halves padding -> spreads LDS banks

__global__ __launch_bounds__(256)
void k_attn_scores(const float* __restrict__ enc,      // [L*B, H] row-major
                   const float* __restrict__ attnW,    // [H, 2H]
                   const float* __restrict__ qp,       // [B, H] (incl. attn_b)
                   const float* __restrict__ scoreW,   // [H]
                   const unsigned char* __restrict__ mask, // [B, L] bool
                   float* __restrict__ scores)         // [B, L]
{
  __shared__ _Float16 As[32 * ASTRIDE];
  __shared__ float rowsum[32];

  const int m0 = blockIdx.x << 5;  // 32-row block
  const int t  = threadIdx.x;
  if (t < 32) rowsum[t] = 0.0f;

  // Stage A: 32 rows x 512 K, fp32 -> f16, each thread 8 elems/iter
  for (int idx = t; idx < 2048; idx += 256) {
    int r = idx >> 6;
    int c = (idx & 63) << 3;
    const float4* g = reinterpret_cast<const float4*>(enc + (size_t)(m0 + r) * HH + c);
    float4 a = g[0], b = g[1];
    v8h h;
    h[0]=(_Float16)a.x; h[1]=(_Float16)a.y; h[2]=(_Float16)a.z; h[3]=(_Float16)a.w;
    h[4]=(_Float16)b.x; h[5]=(_Float16)b.y; h[6]=(_Float16)b.z; h[7]=(_Float16)b.w;
    *reinterpret_cast<v8h*>(&As[r * ASTRIDE + c]) = h;
  }
  __syncthreads();

  const int w    = t >> 5;
  const int lane = t & 31;
  const int half = lane >> 4;
  const int l16  = lane & 15;
  const int akk  = half << 3;   // A lane K offset (0 / 8)
  const int bkk  = half << 3;   // B lane K offset (0 / 8)

  v8f acc[2][4] = {};

  for (int kb = 0; kb < HH; kb += 32) {
    v16h bf[4];
#pragma unroll
    for (int j = 0; j < 4; ++j) {
      const float* wr = attnW + (size_t)(((w << 2) + j) * 16 + l16) * (2 * HH)
                        + HH + kb + bkk;
      load8f(wr,     bf[j], 0);
      load8f(wr + 16, bf[j], 8);
    }
#pragma unroll
    for (int mt = 0; mt < 2; ++mt) {
      const _Float16* ap = &As[(mt * 16 + l16) * ASTRIDE + kb + akk];
      v8h lo = *reinterpret_cast<const v8h*>(ap);
      v8h hi = *reinterpret_cast<const v8h*>(ap + 16);
      v16h af;
#pragma unroll
      for (int i = 0; i < 8; ++i) { af[i] = lo[i]; af[i + 8] = hi[i]; }
#pragma unroll
      for (int j = 0; j < 4; ++j) acc[mt][j] = wmma16(af, bf[j], acc[mt][j]);
    }
  }

  // Epilogue: tanh + dot(score_W), accumulate per-row partials in each lane
  float part[2][8];
#pragma unroll
  for (int mt = 0; mt < 2; ++mt)
#pragma unroll
    for (int v = 0; v < 8; ++v) part[mt][v] = 0.0f;

#pragma unroll
  for (int mt = 0; mt < 2; ++mt) {
#pragma unroll
    for (int j = 0; j < 4; ++j) {
      int n = ((w << 2) + j) * 16 + l16;
      float sw = scoreW[n];
#pragma unroll
      for (int v = 0; v < 8; ++v) {
        int r = m0 + mt * 16 + v + (half << 3);
        int b = r & (BB - 1);
        float val = acc[mt][j][v] + qp[b * HH + n];
        part[mt][v] += sw * fast_tanh(val);
      }
    }
  }

  // Reduce across the 16 lanes of each half-wave (same row, different N)
#pragma unroll
  for (int mt = 0; mt < 2; ++mt) {
#pragma unroll
    for (int v = 0; v < 8; ++v) {
      float p = part[mt][v];
      p += __shfl_xor(p, 1, 32);
      p += __shfl_xor(p, 2, 32);
      p += __shfl_xor(p, 4, 32);
      p += __shfl_xor(p, 8, 32);
      if (l16 == 0) atomicAdd(&rowsum[mt * 16 + v + (half << 3)], p);
    }
  }
  __syncthreads();

  if (t < 32) {
    int r = m0 + t;
    int b = r & (BB - 1);
    int l = r >> 7;               // r / B
    float s = rowsum[t];
    if (mask[b * LL + l]) s = -1e12f;
    scores[b * LL + l] = s;
  }
}

// ---------------------------------------------------------------------------
// Softmax over L per batch row.  Block = 256 threads, each owns 2 elements.
// ---------------------------------------------------------------------------
__global__ void k_softmax(const float* __restrict__ scores, float* __restrict__ aw) {
  __shared__ float red[256];
  const int b = blockIdx.x, t = threadIdx.x;
  const float* srow = scores + b * LL;
  float v0 = srow[t], v1 = srow[t + 256];
  red[t] = fmaxf(v0, v1);
  __syncthreads();
  for (int s = 128; s > 0; s >>= 1) { if (t < s) red[t] = fmaxf(red[t], red[t + s]); __syncthreads(); }
  float mx = red[0];
  __syncthreads();
  float e0 = __expf(v0 - mx), e1 = __expf(v1 - mx);
  red[t] = e0 + e1;
  __syncthreads();
  for (int s = 128; s > 0; s >>= 1) { if (t < s) red[t] += red[t + s]; __syncthreads(); }
  float inv = 1.0f / red[0];
  aw[b * LL + t] = e0 * inv;
  aw[b * LL + t + 256] = e1 * inv;
}

// context[b,h] = sum_l aw[b,l] * enc[l,b,h]
__global__ void k_context(const float* __restrict__ aw, const float* __restrict__ enc,
                          float* __restrict__ ctx) {
  __shared__ float w[LL];
  const int b = blockIdx.x, h = threadIdx.x;   // 512 threads
  w[h] = aw[b * LL + h];
  __syncthreads();
  float acc = 0.0f;
  const float* e = enc + (size_t)b * HH + h;
  for (int l = 0; l < LL; ++l) {
    if (l + 8 < LL) __builtin_prefetch(e + (size_t)(l + 8) * (BB * HH), 0, 1);
    acc += w[l] * e[(size_t)l * (BB * HH)];
  }
  ctx[b * HH + h] = acc;
}

// x0[b] = [ word_dict_vec[b, ids[b], :] | context[b,:] ]
__global__ void k_x0(const int* __restrict__ ids, const float* __restrict__ wdv,
                     const float* __restrict__ ctx, float* __restrict__ x0) {
  const int b = blockIdx.x, h = threadIdx.x;   // 512 threads
  int id = ids[b];
  x0[b * (2 * HH) + h]      = wdv[((size_t)b * VV + id) * HH + h];
  x0[b * (2 * HH) + HH + h] = ctx[b * HH + h];
}

// GRU cell elementwise; gates ordered r,z,n (torch)
__global__ void k_gru(const float* __restrict__ gi, const float* __restrict__ gh,
                      const float* __restrict__ hprev,
                      float* __restrict__ hout, float* __restrict__ hout2) {
  const int i = blockIdx.x * blockDim.x + threadIdx.x;  // B*H
  const int b = i >> 9, h = i & (HH - 1);
  const float* gib = gi + b * (3 * HH);
  const float* ghb = gh + b * (3 * HH);
  float ir = gib[h], iz = gib[HH + h], in_ = gib[2 * HH + h];
  float hr = ghb[h], hz = ghb[HH + h], hn  = ghb[2 * HH + h];
  float r = fast_sigmoid(ir + hr);
  float z = fast_sigmoid(iz + hz);
  float n = fast_tanh(in_ + r * hn);
  float out = (1.0f - z) * n + z * hprev[i];
  hout[i]  = out;
  hout2[i] = out;
}

// cat2[b] = [ h1 | context ]
__global__ void k_cat2(const float* __restrict__ h1, const float* __restrict__ ctx,
                       float* __restrict__ cat2) {
  const int b = blockIdx.x, h = threadIdx.x;
  cat2[b * (2 * HH) + h]      = h1[b * HH + h];
  cat2[b * (2 * HH) + HH + h] = ctx[b * HH + h];
}

// ---------------------------------------------------------------------------
extern "C" void kernel_launch(void* const* d_in, const int* in_sizes, int n_in,
                              void* d_out, int out_size, void* d_ws, size_t ws_size,
                              hipStream_t stream) {
  const int*   ids   = (const int*)  d_in[0];
  const float* lastH = (const float*)d_in[1];   // [2,B,H]
  const float* wdv   = (const float*)d_in[2];   // [B,V,H]
  const float* enc   = (const float*)d_in[3];   // [L,B,H]
  const unsigned char* mask = (const unsigned char*)d_in[4];  // [B,L] bool
  const float* attnW = (const float*)d_in[5];   // [H,2H]
  const float* attnB = (const float*)d_in[6];   // [H]
  const float* scW   = (const float*)d_in[7];   // [1,H]
  const float* catW  = (const float*)d_in[8];   // [H,2H]
  const float* catB  = (const float*)d_in[9];
  const float* Wih0  = (const float*)d_in[10];  // [3H,2H]
  const float* Whh0  = (const float*)d_in[11];  // [3H,H]
  const float* bih0  = (const float*)d_in[12];
  const float* bhh0  = (const float*)d_in[13];
  const float* Wih1  = (const float*)d_in[14];  // [3H,H]
  const float* Whh1  = (const float*)d_in[15];
  const float* bih1  = (const float*)d_in[16];
  const float* bhh1  = (const float*)d_in[17];

  float* out = (float*)d_out;                   // [B,H] output, then [2,B,H] hidden

  float* ws = (float*)d_ws;
  float* qp     = ws;                 ws += BB * HH;        // 65536
  float* scores = ws;                 ws += BB * LL;
  float* aw     = ws;                 ws += BB * LL;
  float* ctx    = ws;                 ws += BB * HH;
  float* x0     = ws;                 ws += BB * 2 * HH;    // 131072
  float* gi0    = ws;                 ws += BB * 3 * HH;    // 196608
  float* gh0    = ws;                 ws += BB * 3 * HH;
  float* h0w    = ws;                 ws += BB * HH;
  float* gi1    = ws;                 ws += BB * 3 * HH;
  float* gh1    = ws;                 ws += BB * 3 * HH;
  float* h1w    = ws;                 ws += BB * HH;
  float* cat2   = ws;                 ws += BB * 2 * HH;

  const float* h_prev0 = lastH;             // layer 0
  const float* h_prev1 = lastH + BB * HH;   // layer 1 ( == q )

  // 1) qp = q @ attn_W[:, :H].T + attn_b     (M=128,N=512,K=512 -> 256 tiles)
  k_gemm<<<64, 128, 0, stream>>>(h_prev1, HH, attnW, 2 * HH, attnB,
                                 qp, HH, BB, HH, HH, 0);
  // 2) fused energy GEMM + tanh + score dot + mask  (65536/32 blocks)
  k_attn_scores<<<2048, 256, 0, stream>>>(enc, attnW, qp, scW, mask, scores);
  // 3) softmax over L
  k_softmax<<<BB, 256, 0, stream>>>(scores, aw);
  // 4) context = aw . enc
  k_context<<<BB, HH, 0, stream>>>(aw, enc, ctx);
  // 5) x0 = [emb | ctx]
  k_x0<<<BB, HH, 0, stream>>>(ids, wdv, ctx, x0);
  // 6) gi0 = x0 @ W_ih0.T + b_ih0   (M=128,N=1536,K=1024 -> 768 tiles)
  k_gemm<<<192, 128, 0, stream>>>(x0, 2 * HH, Wih0, 2 * HH, bih0,
                                  gi0, 3 * HH, BB, 3 * HH, 2 * HH, 0);
  // 7) gh0 = h_prev0 @ W_hh0.T + b_hh0
  k_gemm<<<192, 128, 0, stream>>>(h_prev0, HH, Whh0, HH, bhh0,
                                  gh0, 3 * HH, BB, 3 * HH, HH, 0);
  // 8) h0
  k_gru<<<256, 256, 0, stream>>>(gi0, gh0, h_prev0, h0w, out + BB * HH);
  // 9) gi1 = h0 @ W_ih1.T + b_ih1
  k_gemm<<<192, 128, 0, stream>>>(h0w, HH, Wih1, HH, bih1,
                                  gi1, 3 * HH, BB, 3 * HH, HH, 0);
  // 10) gh1 = h_prev1 @ W_hh1.T + b_hh1
  k_gemm<<<192, 128, 0, stream>>>(h_prev1, HH, Whh1, HH, bhh1,
                                  gh1, 3 * HH, BB, 3 * HH, HH, 0);
  // 11) h1
  k_gru<<<256, 256, 0, stream>>>(gi1, gh1, h_prev1, h1w, out + 2 * BB * HH);
  // 12) cat2 = [h1 | ctx]
  k_cat2<<<BB, HH, 0, stream>>>(h1w, ctx, cat2);
  // 13) output = tanh(cat2 @ concat_W.T + concat_b)
  k_gemm<<<64, 128, 0, stream>>>(cat2, 2 * HH, catW, 2 * HH, catB,
                                 out, HH, BB, HH, 2 * HH, 1);
}
